// EdgeDense_51075751084150
// MI455X (gfx1250) — matrix operand order repaired and checked by reading, not compiled, not run on hardware
//
#include <hip/hip_runtime.h>
#include <hip/hip_bf16.h>

// ---------------------------------------------------------------------------
// EdgeDense: z = x @ W + b  (GEMM, bf16 WMMA)  then  out[e] = a_e*(z[src]+z[dst])
// MI455X / gfx1250, wave32.
//
// Roofline: z (25.6 MB) << L2 (192 MB) -> edge gathers are L2 hits.
// HBM floor ~ 102 MB (x, NT loads) + 410 MB (out, NT stores) + 19 MB (edges).
// GEMM (3.28 GFLOP) runs on the WMMA pipe so it hides under the memory time.
// ---------------------------------------------------------------------------

typedef __attribute__((ext_vector_type(16))) __bf16 v16bf;
typedef __attribute__((ext_vector_type(8)))  __bf16 v8bf;
typedef __attribute__((ext_vector_type(8)))  float  v8f;
typedef __attribute__((ext_vector_type(4)))  float  fvec4;

#define IN_F  256
#define OUT_F 64

// ---------------------------------------------------------------------------
// Kernel 0: W [256,64] f32 (row-major)  ->  W_t [64,256] bf16 (row-major)
// so each WMMA B-fragment is two contiguous 16B runs of K for one output column.
// ---------------------------------------------------------------------------
__global__ void prep_w_kernel(const float* __restrict__ W, __bf16* __restrict__ Wt) {
    int t = blockIdx.x * blockDim.x + threadIdx.x;   // 0 .. 16383
    if (t < IN_F * OUT_F) {
        int k = t >> 6;      // 0..255
        int n = t & 63;      // 0..63
        Wt[n * IN_F + k] = (__bf16)W[k * OUT_F + n];
    }
}

// ---------------------------------------------------------------------------
// Kernel 1: z[m][n] = sum_k x[m][k] * W[k][n] + b[n]
// One wave computes one 16x16 tile of z via 8x V_WMMA_F32_16X16X32_BF16.
// Block = 256 threads = 8 waves = 2 M-tiles x 4 N-tiles.
// x is read once -> non-temporal loads keep it from polluting L2.
// ---------------------------------------------------------------------------
__global__ void __launch_bounds__(256)
gemm_wmma_kernel(const float* __restrict__ x,
                 const __bf16* __restrict__ Wt,
                 const float* __restrict__ bias,
                 float* __restrict__ z,
                 int n_rows) {
    const int lane  = threadIdx.x & 31;
    const int wave  = threadIdx.x >> 5;              // 0..7
    const int mtile = blockIdx.x * 2 + (wave >> 2);
    const int ntile = wave & 3;

    if (mtile * 16 >= n_rows) return;                // full wave uniform exit

    const int halfsel = lane >> 4;                   // 0: lanes 0-15, 1: lanes 16-31
    const int mrow = mtile * 16 + (lane & 15);       // A row held by this lane
    const int ncol = ntile * 16 + (lane & 15);       // B/D column held by this lane

    const float*  xrow = x  + (size_t)mrow * IN_F;
    const __bf16* wrow = Wt + (size_t)ncol * IN_F;

    v8f c = {};

#pragma unroll
    for (int k = 0; k < IN_F; k += 32) {
        // ---- A fragment (16x32 bf16, ISA 16-bit A layout):
        // lane half 0 holds K = k+{0..7, 16..23}; half 1 holds K = k+{8..15, 24..31}
        const int ka = k + (halfsel << 3);
        fvec4 a0 = __builtin_nontemporal_load(reinterpret_cast<const fvec4*>(xrow + ka));
        fvec4 a1 = __builtin_nontemporal_load(reinterpret_cast<const fvec4*>(xrow + ka + 4));
        fvec4 a2 = __builtin_nontemporal_load(reinterpret_cast<const fvec4*>(xrow + ka + 16));
        fvec4 a3 = __builtin_nontemporal_load(reinterpret_cast<const fvec4*>(xrow + ka + 20));
        float af[16] = {a0.x, a0.y, a0.z, a0.w, a1.x, a1.y, a1.z, a1.w,
                        a2.x, a2.y, a2.z, a2.w, a3.x, a3.y, a3.z, a3.w};
        v16bf a;
#pragma unroll
        for (int i = 0; i < 16; ++i) a[i] = (__bf16)af[i];

        // ---- B fragment (32x16 bf16): lane = column, same K order as A:
        // K = k + 8*halfsel + {0..7, 16..23}  (two 16B vector loads; Wt is hot in cache)
        const int kb = k + (halfsel << 3);
        v8bf b0 = *reinterpret_cast<const v8bf*>(wrow + kb);
        v8bf b1 = *reinterpret_cast<const v8bf*>(wrow + kb + 16);
        v16bf b;
#pragma unroll
        for (int i = 0; i < 8; ++i) { b[i] = b0[i]; b[8 + i] = b1[i]; }

        // D = A*B + C   (8 args: neg_a, A, neg_b, B, c_mod, C, reuse_a, reuse_b)
        c = __builtin_amdgcn_wmma_f32_16x16x32_bf16(false, a, false, b,
                                                    (short)0, c, false, false);
    }

    const float bv = bias[ncol];

    // C/D layout: VGPR r -> row (r + 8*halfsel), column = lane&15
#pragma unroll
    for (int r = 0; r < 8; ++r) {
        int m = mtile * 16 + r + (halfsel << 3);
        z[(size_t)m * OUT_F + ncol] = c[r] + bv;
    }
}

// ---------------------------------------------------------------------------
// Kernel 2: out[e][:] = adj[e] * (z[src[e]][:] + z[dst[e]][:])
// 16 threads per edge, float4 per thread -> coalesced b128 traffic.
// z gathers use default (RT) policy -> L2 hits; out/adj/src/dst are streamed
// non-temporally so the 410 MB output stream does not evict z from L2.
// ---------------------------------------------------------------------------
__global__ void __launch_bounds__(256)
edge_kernel(const float* __restrict__ z,
            const float* __restrict__ adj,
            const int* __restrict__ src,
            const int* __restrict__ dst,
            float* __restrict__ out,
            int n_edges) {
    int t  = blockIdx.x * blockDim.x + threadIdx.x;
    int e  = t >> 4;           // edge index
    int f4 = t & 15;           // float4 chunk within the 64 features
    if (e >= n_edges) return;

    int   s = __builtin_nontemporal_load(src + e);
    int   d = __builtin_nontemporal_load(dst + e);
    float a = __builtin_nontemporal_load(adj + e);

    const fvec4* z4 = reinterpret_cast<const fvec4*>(z);
    fvec4 zs = z4[(size_t)s * 16 + f4];      // RT: keep z resident in L2
    fvec4 zd = z4[(size_t)d * 16 + f4];

    fvec4 r = a * (zs + zd);

    __builtin_nontemporal_store(r, reinterpret_cast<fvec4*>(out) + (size_t)e * 16 + f4);
}

// ---------------------------------------------------------------------------
// Launch
// inputs: 0:x [N,256] f32   1:W [256,64] f32   2:b [64] f32
//         3:adj_vals [E] f32   4:src [E] int   5:dst [E] int
// ws layout: [0, 32KB)   W_t bf16 [64][256]
//            [64KB, ...) z   f32  [N][64]
// ---------------------------------------------------------------------------
extern "C" void kernel_launch(void* const* d_in, const int* in_sizes, int n_in,
                              void* d_out, int out_size, void* d_ws, size_t ws_size,
                              hipStream_t stream) {
    const float* x    = (const float*)d_in[0];
    const float* W    = (const float*)d_in[1];
    const float* bias = (const float*)d_in[2];
    const float* adj  = (const float*)d_in[3];
    const int*   src  = (const int*)d_in[4];
    const int*   dst  = (const int*)d_in[5];
    float*       out  = (float*)d_out;

    const int n_rows  = in_sizes[0] / IN_F;   // 100000
    const int n_edges = in_sizes[3];          // 1600000

    __bf16* Wt = (__bf16*)d_ws;
    float*  z  = (float*)((char*)d_ws + 65536);

    // 0) transpose/convert W
    prep_w_kernel<<<(IN_F * OUT_F + 255) / 256, 256, 0, stream>>>(W, Wt);

    // 1) GEMM: 2 M-tiles (32 rows) per block
    int gemm_blocks = (n_rows + 31) / 32;
    gemm_wmma_kernel<<<gemm_blocks, 256, 0, stream>>>(x, Wt, bias, z, n_rows);

    // 2) edge gather/scale: 16 edges per 256-thread block
    long long work = (long long)n_edges * 16;
    int edge_blocks = (int)((work + 255) / 256);
    edge_kernel<<<edge_blocks, 256, 0, stream>>>(z, adj, src, dst, out, n_edges);
}
